// GNNPolicy_11802570130124
// MI455X (gfx1250) — compile-verified, matrix operand order checked by reading, not run on hardware
//
#include <hip/hip_runtime.h>
#include <hip/hip_bf16.h>

typedef __attribute__((ext_vector_type(16))) _Float16 v16h;
typedef __attribute__((ext_vector_type(8)))  _Float16 v8h;
typedef __attribute__((ext_vector_type(8)))  float    v8f;

#define HIDN 128

// ---------- helpers: order-preserving float<->uint keys for atomic max ----------
__device__ __forceinline__ unsigned fkey(float f) {
    unsigned u = __float_as_uint(f);
    return (u >> 31) ? ~u : (u | 0x80000000u);
}
__device__ __forceinline__ float kfloat(unsigned k) {
    return __uint_as_float((k >> 31) ? (k & 0x7fffffffu) : ~k);
}

// ---------- degree over src ----------
__global__ void k_deg_src(const long long* __restrict__ ei, float* __restrict__ deg, long long E) {
    long long e = (long long)blockIdx.x * blockDim.x + threadIdx.x;
    if (e < E) atomicAdd(deg + ei[e], 1.0f);
}

// ---------- block-reduced keyed max -> cell ----------
__global__ void k_maxkey(const float* __restrict__ v, unsigned* __restrict__ cell, int n) {
    __shared__ unsigned sm[256];
    int i = blockIdx.x * 256 + threadIdx.x;
    sm[threadIdx.x] = (i < n) ? fkey(v[i]) : 0u;
    __syncthreads();
    for (int s = 128; s > 0; s >>= 1) {
        if (threadIdx.x < (unsigned)s) sm[threadIdx.x] = max(sm[threadIdx.x], sm[threadIdx.x + s]);
        __syncthreads();
    }
    if (threadIdx.x == 0) atomicMax(cell, sm[0]);
}

// ---------- features [coords | deg/max(deg)] ; also init deg_hat = 1 (self loop) ----------
__global__ void k_feat(const float* __restrict__ coords, const float* __restrict__ deg,
                       const unsigned* __restrict__ dmaxkey,
                       float* __restrict__ feat, float* __restrict__ deg_hat, int n) {
    int i = blockIdx.x * blockDim.x + threadIdx.x;
    if (i >= n) return;
    float dm = fmaxf(kfloat(*dmaxkey), 1.0f);
    feat[i * 3 + 0] = coords[i * 2 + 0];
    feat[i * 3 + 1] = coords[i * 2 + 1];
    feat[i * 3 + 2] = deg[i] / dm;
    deg_hat[i] = 1.0f;   // self loop
}

// ---------- deg_hat += in-degree(dst) ----------
__global__ void k_deg_dst(const long long* __restrict__ ei, float* __restrict__ deg_hat, long long E) {
    long long e = (long long)blockIdx.x * blockDim.x + threadIdx.x;
    if (e < E) atomicAdd(deg_hat + ei[E + e], 1.0f);
}

__global__ void k_dinv(float* __restrict__ d, int n) {
    int i = blockIdx.x * blockDim.x + threadIdx.x;
    if (i < n) d[i] = rsqrtf(d[i]);   // deg_hat >= 1 always
}

// ---------- layer-1 transform: h = feat(Nx3) @ W1(3x128) ----------
__global__ void k_h1(const float* __restrict__ feat, const float* __restrict__ W1,
                     float* __restrict__ h, int n) {
    int t = blockIdx.x * blockDim.x + threadIdx.x;
    if (t >= n * HIDN) return;
    int i = t >> 7, j = t & 127;
    const float* f = feat + i * 3;
    h[t] = f[0] * W1[0 * HIDN + j] + f[1] * W1[1 * HIDN + j] + f[2] * W1[2 * HIDN + j];
}

// ---------- edge aggregation: acc[d] += h[s] * dinv[s]*dinv[d]  (32 lanes / edge) ----------
__global__ __launch_bounds__(256) void k_agg(const long long* __restrict__ ei,
                                             const float* __restrict__ h,
                                             const float* __restrict__ dinv,
                                             float* __restrict__ acc, long long E) {
    long long e = (long long)blockIdx.x * 8 + (threadIdx.x >> 5);
    int lane = threadIdx.x & 31;
    if (e >= E) return;
    long long s = ei[e], d = ei[E + e];
    float w = dinv[s] * dinv[d];
    const float4 hv = *(const float4*)(h + s * HIDN + lane * 4);
    float* ap = acc + d * HIDN + lane * 4;
    atomicAdd(ap + 0, hv.x * w);
    atomicAdd(ap + 1, hv.y * w);
    atomicAdd(ap + 2, hv.z * w);
    atomicAdd(ap + 3, hv.w * w);
}

// ---------- finalize: x = relu(acc + h*dinv^2 + b) (in place in acc) ----------
__global__ void k_finalize(float* __restrict__ acc, const float* __restrict__ h,
                           const float* __restrict__ dinv, const float* __restrict__ b, int n) {
    int t = blockIdx.x * blockDim.x + threadIdx.x;
    if (t >= n * HIDN) return;
    int i = t >> 7, j = t & 127;
    float di = dinv[i];
    float v = acc[t] + h[t] * di * di + b[j];
    acc[t] = fmaxf(v, 0.0f);
}

// ---------- f32 -> f16 convert ----------
__global__ void k_cvt16(const float* __restrict__ x, _Float16* __restrict__ y, int nelem) {
    int t = blockIdx.x * blockDim.x + threadIdx.x;
    if (t < nelem) y[t] = (_Float16)x[t];
}

// ---------- pack W2 transposed to f16: Bt[n][k] = W2[k][n] ----------
__global__ void k_packW(const float* __restrict__ W2, _Float16* __restrict__ Bt) {
    int t = blockIdx.x * blockDim.x + threadIdx.x;   // t < 128*128
    int nn = t >> 7, k = t & 127;
    Bt[nn * HIDN + k] = (_Float16)W2[k * HIDN + nn];
}

// ---------- WMMA GEMM: C(N x128) = A(N x128,f16) @ B(128x128, Bt f16) ----------
__global__ __launch_bounds__(256) void k_gemm_wmma(const _Float16* __restrict__ A,
                                                   const _Float16* __restrict__ Bt,
                                                   float* __restrict__ C, int nrows) {
    const int wave = threadIdx.x >> 5;        // 0..7 -> 16-row strip
    const int lane = threadIdx.x & 31;
    const int half = lane >> 4;               // K-half select per ISA layout
    const int l16  = lane & 15;
    const int rowBase = blockIdx.x * 128 + wave * 16;
    const bool full = (rowBase + 16) <= nrows;   // wave-uniform: no exec divergence

    int arow = rowBase + l16;
    if (arow >= nrows) arow = nrows - 1;      // clamp loads; stores are guarded
    const _Float16* arp = A + (size_t)arow * HIDN;

    // A is streamed exactly once -> non-temporal loads keep L2 for hbuf/W2
    v16h a[4];
    #pragma unroll
    for (int ks = 0; ks < 4; ++ks) {
        const int kb = ks * 32 + half * 8;
        v8h lo = __builtin_nontemporal_load((const v8h*)(arp + kb));
        v8h hi = __builtin_nontemporal_load((const v8h*)(arp + kb + 16));
        #pragma unroll
        for (int q = 0; q < 8; ++q) { a[ks][q] = lo[q]; a[ks][q + 8] = hi[q]; }
    }

    #pragma unroll
    for (int nt = 0; nt < 8; ++nt) {
        const _Float16* brp = Bt + (size_t)(nt * 16 + l16) * HIDN;
        v8f c = {};
        #pragma unroll
        for (int ks = 0; ks < 4; ++ks) {
            const int kb = ks * 32 + half * 8;
            v8h lo = *(const v8h*)(brp + kb);
            v8h hi = *(const v8h*)(brp + kb + 16);
            v16h b;
            #pragma unroll
            for (int q = 0; q < 8; ++q) { b[q] = lo[q]; b[q + 8] = hi[q]; }
            c = __builtin_amdgcn_wmma_f32_16x16x32_f16(false, a[ks], false, b,
                                                       (short)0, c, false, false);
        }
        // C layout: lane -> col, VGPR j -> row (j + 8*half)
        float* cp = C + (size_t)(rowBase + half * 8) * HIDN + nt * 16 + l16;
        if (full) {
            #pragma unroll
            for (int j = 0; j < 8; ++j) cp[(size_t)j * HIDN] = c[j];
        } else {
            #pragma unroll
            for (int j = 0; j < 8; ++j) {
                if (rowBase + half * 8 + j < nrows) cp[(size_t)j * HIDN] = c[j];
            }
        }
    }
}

// ---------- per-node heads: scores, mu, log_std ----------
__global__ void k_heads(const float* __restrict__ x, const float* __restrict__ Ws,
                        const float* __restrict__ bs, const float* __restrict__ Wmu,
                        const float* __restrict__ bmu, const float* __restrict__ Wls,
                        const float* __restrict__ bls, float* __restrict__ scores,
                        float* __restrict__ out, int n) {
    int i = blockIdx.x * blockDim.x + threadIdx.x;
    if (i >= n) return;
    const float* xr = x + (size_t)i * HIDN;
    float s = 0.f, m0 = 0.f, m1 = 0.f, l0 = 0.f, l1 = 0.f;
    for (int k = 0; k < HIDN; ++k) {
        float v = xr[k];
        s  += v * Ws[k];
        m0 += v * Wmu[k * 2 + 0];
        m1 += v * Wmu[k * 2 + 1];
        l0 += v * Wls[k * 2 + 0];
        l1 += v * Wls[k * 2 + 1];
    }
    scores[i] = s + bs[0];
    out[n + i * 2 + 0]     = m0 + bmu[0];
    out[n + i * 2 + 1]     = m1 + bmu[1];
    out[3 * n + i * 2 + 0] = fminf(fmaxf(l0 + bls[0], -4.0f), 2.0f);
    out[3 * n + i * 2 + 1] = fminf(fmaxf(l1 + bls[1], -4.0f), 2.0f);
}

// ---------- column sums of x -> gsum[128] ----------
__global__ __launch_bounds__(256) void k_gsum(const float* __restrict__ x,
                                              float* __restrict__ gsum, int n) {
    int j  = threadIdx.x & 127;
    int i0 = blockIdx.x * 2 + (threadIdx.x >> 7);
    float acc = 0.f;
    for (int i = i0; i < n; i += gridDim.x * 2) acc += x[(size_t)i * HIDN + j];
    atomicAdd(gsum + j, acc);
}

// ---------- value head (single block, 64 threads) ----------
__global__ void k_value(const float* __restrict__ gsum, const float* __restrict__ Wv1,
                        const float* __restrict__ bv1, const float* __restrict__ Wv2,
                        const float* __restrict__ bv2, float* __restrict__ out,
                        float invN, int n) {
    __shared__ float sm[64];
    int t = threadIdx.x;
    float acc = 0.f;
    for (int k = 0; k < HIDN; ++k) acc += (gsum[k] * invN) * Wv1[k * 64 + t];
    sm[t] = fmaxf(acc + bv1[t], 0.0f) * Wv2[t];
    __syncthreads();
    for (int s = 32; s > 0; s >>= 1) {
        if (t < s) sm[t] += sm[t + s];
        __syncthreads();
    }
    if (t == 0) out[5 * (size_t)n] = sm[0] + bv2[0];
}

// ---------- log-softmax pieces ----------
__global__ void k_sumexp(const float* __restrict__ s, const unsigned* __restrict__ maxkey,
                         float* __restrict__ sumcell, int n) {
    __shared__ float sm[256];
    float m = kfloat(*maxkey);
    int i = blockIdx.x * 256 + threadIdx.x;
    sm[threadIdx.x] = (i < n) ? expf(s[i] - m) : 0.f;
    __syncthreads();
    for (int st = 128; st > 0; st >>= 1) {
        if (threadIdx.x < (unsigned)st) sm[threadIdx.x] += sm[threadIdx.x + st];
        __syncthreads();
    }
    if (threadIdx.x == 0) atomicAdd(sumcell, sm[0]);
}

__global__ void k_logits(const float* __restrict__ s, const unsigned* __restrict__ maxkey,
                         const float* __restrict__ sumcell, float* __restrict__ out, int n) {
    int i = blockIdx.x * blockDim.x + threadIdx.x;
    if (i < n) out[i] = s[i] - kfloat(*maxkey) - logf(*sumcell);
}

extern "C" void kernel_launch(void* const* d_in, const int* in_sizes, int n_in,
                              void* d_out, int out_size, void* d_ws, size_t ws_size,
                              hipStream_t stream) {
    const float*     coords = (const float*)d_in[0];
    const long long* ei     = (const long long*)d_in[1];   // int64 per reference
    const float* W1  = (const float*)d_in[2];
    const float* b1  = (const float*)d_in[3];
    const float* W2  = (const float*)d_in[4];
    const float* b2  = (const float*)d_in[5];
    const float* Ws  = (const float*)d_in[6];
    const float* bs  = (const float*)d_in[7];
    const float* Wmu = (const float*)d_in[8];
    const float* bmu = (const float*)d_in[9];
    const float* Wls = (const float*)d_in[10];
    const float* bls = (const float*)d_in[11];
    const float* Wv1 = (const float*)d_in[12];
    const float* bv1 = (const float*)d_in[13];
    const float* Wv2 = (const float*)d_in[14];
    const float* bv2 = (const float*)d_in[15];
    float* out = (float*)d_out;

    const int       N = in_sizes[0] / 2;
    const long long E = (long long)in_sizes[1] / 2;

    // workspace layout (float units, 16B-aligned regions)
    float* ws = (float*)d_ws;
    size_t off = 0;
    auto take = [&](size_t nf) { size_t o = off; off += (nf + 3) & ~(size_t)3; return o; };
    size_t o_degsrc = take(N);            // degree over src
    size_t o_dinv   = take(N);            // deg_hat then rsqrt in place
    size_t o_scal   = take(16);           // [0]=dmax key, [1]=smax key, [2]=ssum
    size_t o_gsum   = take(HIDN);         // column sums for mean
    size_t o_feat   = take((size_t)N * 3);
    size_t o_h      = take((size_t)N * HIDN);   // transform output (both layers)
    size_t o_acc    = take((size_t)N * HIDN);   // aggregation acc -> layer output
    size_t o_ah     = take((size_t)N * HIDN / 2); // f16 copy of x1
    size_t o_scores = take(N);
    size_t o_w2h    = take((size_t)HIDN * HIDN / 2);
    (void)ws_size;

    float*     deg_src = ws + o_degsrc;
    float*     dinv    = ws + o_dinv;
    unsigned*  dmaxkey = (unsigned*)(ws + o_scal) + 0;
    unsigned*  smaxkey = (unsigned*)(ws + o_scal) + 1;
    float*     ssum    = ws + o_scal + 2;
    float*     gsum    = ws + o_gsum;
    float*     feat    = ws + o_feat;
    float*     hbuf    = ws + o_h;
    float*     acc     = ws + o_acc;
    _Float16*  ah      = (_Float16*)(ws + o_ah);
    float*     scores  = ws + o_scores;
    _Float16*  w2h     = (_Float16*)(ws + o_w2h);

    const int  TB   = 256;
    const int  gN   = (N + TB - 1) / TB;
    const int  gNH  = (int)(((size_t)N * HIDN + TB - 1) / TB);
    const int  gE   = (int)((E + TB - 1) / TB);
    const int  gAgg = (int)((E + 7) / 8);          // 32 lanes per edge
    const int  gGemm = (N + 127) / 128;

    // ---- degree feature ----
    hipMemsetAsync(deg_src, 0, (size_t)N * 4, stream);
    hipMemsetAsync(ws + o_scal, 0, (16 + HIDN + 3) * 4, stream);  // keys, ssum, gsum
    k_deg_src<<<gE, TB, 0, stream>>>(ei, deg_src, E);
    k_maxkey<<<gN, TB, 0, stream>>>(deg_src, dmaxkey, N);
    k_feat<<<gN, TB, 0, stream>>>(coords, deg_src, dmaxkey, feat, dinv, N);

    // ---- GCN normalization (same for both layers) ----
    k_deg_dst<<<gE, TB, 0, stream>>>(ei, dinv, E);
    k_dinv<<<gN, TB, 0, stream>>>(dinv, N);

    // ---- layer 1 ----
    k_h1<<<gNH, TB, 0, stream>>>(feat, W1, hbuf, N);
    hipMemsetAsync(acc, 0, (size_t)N * HIDN * 4, stream);
    k_agg<<<gAgg, TB, 0, stream>>>(ei, hbuf, dinv, acc, E);
    k_finalize<<<gNH, TB, 0, stream>>>(acc, hbuf, dinv, b1, N);   // acc = x1

    // ---- layer 2: WMMA transform then aggregate ----
    k_cvt16<<<gNH, TB, 0, stream>>>(acc, ah, N * HIDN);
    k_packW<<<(HIDN * HIDN) / TB, TB, 0, stream>>>(W2, w2h);
    k_gemm_wmma<<<gGemm, TB, 0, stream>>>(ah, w2h, hbuf, N);      // hbuf = x1 @ W2
    hipMemsetAsync(acc, 0, (size_t)N * HIDN * 4, stream);
    k_agg<<<gAgg, TB, 0, stream>>>(ei, hbuf, dinv, acc, E);
    k_finalize<<<gNH, TB, 0, stream>>>(acc, hbuf, dinv, b2, N);   // acc = x2

    // ---- heads ----
    k_heads<<<gN, TB, 0, stream>>>(acc, Ws, bs, Wmu, bmu, Wls, bls, scores, out, N);
    k_gsum<<<1024, TB, 0, stream>>>(acc, gsum, N);
    k_value<<<1, 64, 0, stream>>>(gsum, Wv1, bv1, Wv2, bv2, out, 1.0f / (float)N, N);
    k_maxkey<<<gN, TB, 0, stream>>>(scores, smaxkey, N);
    k_sumexp<<<gN, TB, 0, stream>>>(scores, smaxkey, ssum, N);
    k_logits<<<gN, TB, 0, stream>>>(scores, smaxkey, ssum, out, N);
}
